// miniLDB_80831284511345
// MI455X (gfx1250) — matrix operand order repaired and checked by reading, not compiled, not run on hardware
//
#include <hip/hip_runtime.h>
#include <hip/hip_bf16.h>
#include <math.h>

// ---------------------------------------------------------------------------
// CDNA5 (gfx1250) implementation: all three GEMM stages run on
// v_wmma_f32_16x16x32_bf16 (wave32). Weights are repacked once into
// A-fragment order (bf16) so each lane's A operand is a single contiguous
// 32B load. K is reordered tap-major so each 32-wide K slice is one 3x3 tap
// x 32 channels. All gathers are branch-free (clamped address + zero scale)
// and grouped into clause-able load runs feeding a single wait.
// ---------------------------------------------------------------------------

typedef __attribute__((ext_vector_type(16))) __bf16 v16bf;
typedef __attribute__((ext_vector_type(8)))  float  v8f;

#define HDIM 128
#define WDIM 128
#define HW   16384
#define CDIM 64

// A-fragment k order used here (B matches): lane<16 element j -> K = j
//                               lane>=16 element j -> K = j+16
__device__ __forceinline__ int koffA(int lane, int j) {
    return (lane < 16) ? j : j + 16;
}

__device__ __forceinline__ v16bf load_afrag(const __bf16* p, int kb, int MT, int mt, int lane) {
    return *(const v16bf*)(p + (((size_t)kb * MT + mt) * 32 + lane) * 16);
}

// ---------------------------------------------------------------------------
// Pack kernel: rewrite weights into fragment-ordered bf16.
//   pw_off : M=32 (27 padded), K=1152, K = tap*128 + cin   (cin: 0-63 x, 64-127 inter)
//   pw_dcn : M=64, K=576,  K = tap*64 + c
//   pg/pb  : M=64, K=64
// dst[e], e = ((kb*MT + mt)*32 + lane)*16 + j
// ---------------------------------------------------------------------------
__global__ void k_pack(const float* __restrict__ w_off, const float* __restrict__ w_dcn,
                       const float* __restrict__ g1, const float* __restrict__ g2,
                       const float* __restrict__ b1, const float* __restrict__ b2,
                       __bf16* __restrict__ pw_off, __bf16* __restrict__ pw_dcn,
                       __bf16* __restrict__ pg1, __bf16* __restrict__ pg2,
                       __bf16* __restrict__ pb1, __bf16* __restrict__ pb2) {
    const int tid  = blockIdx.x * blockDim.x + threadIdx.x;
    const int nthr = gridDim.x * blockDim.x;

    // offset-conv weights: KB=36, MT=2
    for (int e = tid; e < 36 * 2 * 512; e += nthr) {
        const int j = e & 15, lane = (e >> 4) & 31, mt = (e >> 9) & 1, kb = e >> 10;
        const int M = mt * 16 + (lane & 15);
        const int K = kb * 32 + koffA(lane, j);
        float v = 0.f;
        if (M < 27) {
            const int t = K >> 7, cin = K & 127;             // (27,128,3,3)
            v = w_off[(size_t)(M * 128 + cin) * 9 + t];
        }
        pw_off[e] = (__bf16)v;
    }
    // dcn weights: KB=18, MT=4
    for (int e = tid; e < 18 * 4 * 512; e += nthr) {
        const int j = e & 15, lane = (e >> 4) & 31, mt = (e >> 9) & 3, kb = e >> 11;
        const int M = mt * 16 + (lane & 15);
        const int K = kb * 32 + koffA(lane, j);
        const int t = K >> 6, c = K & 63;                     // (64,64,3,3)
        pw_dcn[e] = (__bf16)w_dcn[(size_t)(M * 64 + c) * 9 + t];
    }
    // 1x1 (SFT) weights: KB=2, MT=4
    for (int e = tid; e < 2 * 4 * 512; e += nthr) {
        const int j = e & 15, lane = (e >> 4) & 31, mt = (e >> 9) & 3, kb = e >> 11;
        const int M = mt * 16 + (lane & 15);
        const int K = kb * 32 + koffA(lane, j);
        pg1[e] = (__bf16)g1[M * 64 + K];
        pg2[e] = (__bf16)g2[M * 64 + K];
        pb1[e] = (__bf16)b1[M * 64 + K];
        pb2[e] = (__bf16)b2[M * 64 + K];
    }
}

// ---------------------------------------------------------------------------
// Kernel 1: offset conv. Per wave: 16-pixel tile, 32 output rows (27 valid).
// GEMM M=32, K=1152 (36 K-blocks of tap x 32ch), N=16 per wave.
// Fully unrolled taps; branch-free padded gathers.
// ---------------------------------------------------------------------------
__global__ __launch_bounds__(256)
void k_off_conv(const float* __restrict__ x, const float* __restrict__ inter,
                const __bf16* __restrict__ pw, const float* __restrict__ b_off,
                float* __restrict__ offbuf) {
    const int b     = blockIdx.y;
    const int lane  = threadIdx.x & 31;
    const int wave  = threadIdx.x >> 5;
    const int pbase = (blockIdx.x * 8 + wave) * 16;          // 16 consecutive pixels in a row
    const int h     = pbase >> 7;
    const int w     = (pbase & 127) + (lane & 15);
    const bool hi   = lane >= 16;
    const int csub  = hi ? 16 : 0;
    const float* xb = x     + (size_t)b * CDIM * HW;
    const float* ib = inter + (size_t)b * CDIM * HW;

    v8f acc0 = {}, acc1 = {};
#pragma unroll
    for (int t = 0; t < 9; ++t) {
        const int dh = (t / 3) - 1, dw = (t % 3) - 1;         // compile-time per tap
        const int yy = h + dh;
        const int xx = w + dw;
        const bool valid = (yy >= 0) && (yy < HDIM) && (xx >= 0) && (xx < WDIM);
        const float scale = valid ? 1.f : 0.f;                // zero-pad via scale
        const int sp = min(max(yy, 0), HDIM - 1) * WDIM + min(max(xx, 0), WDIM - 1);
#pragma unroll
        for (int cs = 0; cs < 4; ++cs) {
            // cs 0/1 -> x channels 0..63 ; cs 2/3 -> inter channels 0..63
            const float* bp = ((cs < 2) ? xb : ib) + (size_t)((cs & 1) * 32 + csub) * HW + sp;
            float tmp[16];
#pragma unroll
            for (int j = 0; j < 16; ++j)                      // one clause of 16 loads
                tmp[j] = bp[(size_t)j * HW];
            v16bf bfrag;
#pragma unroll
            for (int j = 0; j < 16; ++j)
                bfrag[j] = (__bf16)(tmp[j] * scale);
            const int kb = t * 4 + cs;
            v16bf a0 = load_afrag(pw, kb, 2, 0, lane);
            v16bf a1 = load_afrag(pw, kb, 2, 1, lane);
            acc0 = __builtin_amdgcn_wmma_f32_16x16x32_bf16(false, a0, false, bfrag, (short)0, acc0, false, false);
            acc1 = __builtin_amdgcn_wmma_f32_16x16x32_bf16(false, a1, false, bfrag, (short)0, acc1, false, false);
        }
    }
    // Epilogue: bias, sigmoid on mask channels (18..26), drop 27..31.
    float* ob = offbuf + (size_t)b * 27 * HW;
    const int pix = h * WDIM + w;
#pragma unroll
    for (int r = 0; r < 8; ++r) {
        const int o0 = hi ? (8 + r) : r;                      // 0..15 -> dy/dx
        ob[(size_t)o0 * HW + pix] = acc0[r] + b_off[o0];
        const int o1 = o0 + 16;                               // 16..31
        if (o1 < 27) {
            float v = acc1[r] + b_off[o1];
            if (o1 >= 18) v = 1.f / (1.f + __expf(-v));       // mask = sigmoid
            ob[(size_t)o1 * HW + pix] = v;
        }
    }
}

// ---------------------------------------------------------------------------
// Kernel 2: fused bilinear sampling + DCN GEMM (M=64, K=576). Writes dcn to out.
// ---------------------------------------------------------------------------
__global__ __launch_bounds__(256)
void k_dcn(const float* __restrict__ x, const float* __restrict__ offbuf,
           const __bf16* __restrict__ pw, float* __restrict__ out) {
    const int b     = blockIdx.y;
    const int lane  = threadIdx.x & 31;
    const int wave  = threadIdx.x >> 5;
    const int pbase = (blockIdx.x * 8 + wave) * 16;
    const int h     = pbase >> 7;
    const int w     = (pbase & 127) + (lane & 15);
    const bool hi   = lane >= 16;
    const int csub  = hi ? 16 : 0;
    const int pix   = h * WDIM + w;
    const float* xb = x      + (size_t)b * CDIM * HW;
    const float* ob = offbuf + (size_t)b * 27 * HW;

    v8f acc[4] = {};
#pragma unroll
    for (int t = 0; t < 9; ++t) {
        const int dh = (t / 3) - 1, dw = (t % 3) - 1;         // compile-time per tap
        const float dy = ob[(size_t)(2 * t) * HW + pix];
        const float dx = ob[(size_t)(2 * t + 1) * HW + pix];
        const float mk = ob[(size_t)(18 + t) * HW + pix];
        const float sy = (float)(h + dh) + dy;
        const float sx = (float)(w + dw) + dx;
        const float y0f = floorf(sy), x0f = floorf(sx);
        const float fy = sy - y0f, fx = sx - x0f;
        const int y0 = (int)y0f, x0 = (int)x0f;
        const int y1 = y0 + 1, x1 = x0 + 1;
        const float vy0 = (y0 >= 0 && y0 < HDIM) ? 1.f : 0.f;
        const float vy1 = (y1 >= 0 && y1 < HDIM) ? 1.f : 0.f;
        const float vx0 = (x0 >= 0 && x0 < WDIM) ? 1.f : 0.f;
        const float vx1 = (x1 >= 0 && x1 < WDIM) ? 1.f : 0.f;
        const int yc0 = min(max(y0, 0), HDIM - 1), yc1 = min(max(y1, 0), HDIM - 1);
        const int xc0 = min(max(x0, 0), WDIM - 1), xc1 = min(max(x1, 0), WDIM - 1);
        const float w00 = (1.f - fy) * (1.f - fx) * vy0 * vx0 * mk;
        const float w01 = (1.f - fy) * fx         * vy0 * vx1 * mk;
        const float w10 = fy * (1.f - fx)         * vy1 * vx0 * mk;
        const float w11 = fy * fx                 * vy1 * vx1 * mk;
        const int sp00 = yc0 * WDIM + xc0, sp01 = yc0 * WDIM + xc1;
        const int sp10 = yc1 * WDIM + xc0, sp11 = yc1 * WDIM + xc1;
#pragma unroll
        for (int cs = 0; cs < 2; ++cs) {
            const float* bp = xb + (size_t)(cs * 32 + csub) * HW;
            float v[16];
#pragma unroll
            for (int j = 0; j < 16; ++j) v[j]  = w00 * bp[(size_t)j * HW + sp00];
#pragma unroll
            for (int j = 0; j < 16; ++j) v[j] += w01 * bp[(size_t)j * HW + sp01];
#pragma unroll
            for (int j = 0; j < 16; ++j) v[j] += w10 * bp[(size_t)j * HW + sp10];
#pragma unroll
            for (int j = 0; j < 16; ++j) v[j] += w11 * bp[(size_t)j * HW + sp11];
            v16bf bfrag;
#pragma unroll
            for (int j = 0; j < 16; ++j) bfrag[j] = (__bf16)v[j];
            const int kb = t * 2 + cs;
#pragma unroll
            for (int mt = 0; mt < 4; ++mt) {
                v16bf a = load_afrag(pw, kb, 4, mt, lane);
                acc[mt] = __builtin_amdgcn_wmma_f32_16x16x32_bf16(false, a, false, bfrag, (short)0, acc[mt], false, false);
            }
        }
    }
    float* outb = out + (size_t)b * CDIM * HW;
#pragma unroll
    for (int mt = 0; mt < 4; ++mt)
#pragma unroll
        for (int r = 0; r < 8; ++r) {
            const int o = mt * 16 + (hi ? 8 + r : r);
            outb[(size_t)o * HW + pix] = acc[mt][r];
        }
}

// ---------------------------------------------------------------------------
// Kernel 3: SFT branches + final combine.
// ---------------------------------------------------------------------------
__device__ __forceinline__ void gemm64(const __bf16* pa, const v16bf bfr[2], int lane, v8f acc[4]) {
#pragma unroll
    for (int kb = 0; kb < 2; ++kb)
#pragma unroll
        for (int mt = 0; mt < 4; ++mt) {
            v16bf a = load_afrag(pa, kb, 4, mt, lane);
            acc[mt] = __builtin_amdgcn_wmma_f32_16x16x32_bf16(false, a, false, bfr[kb], (short)0, acc[mt], false, false);
        }
}

__device__ __forceinline__ void store_h_lrelu(float* swv, const v8f acc[4], int lane) {
    const int col = lane & 15;
    const bool hi = lane >= 16;
#pragma unroll
    for (int mt = 0; mt < 4; ++mt)
#pragma unroll
        for (int r = 0; r < 8; ++r) {
            float v = acc[mt][r];
            v = v >= 0.f ? v : 0.1f * v;                      // leaky relu
            const int row = mt * 16 + (hi ? 8 + r : r);
            swv[row * 16 + col] = v;
        }
}

__device__ __forceinline__ void load_hfrag(const float* swv, v16bf bh[2], int lane) {
    const int col = lane & 15;
#pragma unroll
    for (int kb = 0; kb < 2; ++kb) {
        const int rb = kb * 32 + ((lane >= 16) ? 16 : 0);
        float tmp[16];
#pragma unroll
        for (int j = 0; j < 16; ++j)                          // clause of ds loads
            tmp[j] = swv[(rb + j) * 16 + col];
#pragma unroll
        for (int j = 0; j < 16; ++j)
            bh[kb][j] = (__bf16)tmp[j];
    }
}

__global__ __launch_bounds__(256)
void k_sft_final(const float* __restrict__ x, const float* __restrict__ inter,
                 const __bf16* __restrict__ pg1, const __bf16* __restrict__ pg2,
                 const __bf16* __restrict__ pb1, const __bf16* __restrict__ pb2,
                 float* __restrict__ out) {
    __shared__ float sh[8][CDIM * 16];                        // per-wave transpose buffer
    const int b     = blockIdx.y;
    const int lane  = threadIdx.x & 31;
    const int wave  = threadIdx.x >> 5;
    const int pbase = (blockIdx.x * 8 + wave) * 16;
    const int h     = pbase >> 7;
    const int w     = (pbase & 127) + (lane & 15);
    const bool hi   = lane >= 16;
    const int csub  = hi ? 16 : 0;
    const int pix   = h * WDIM + w;
    const float* xb = x     + (size_t)b * CDIM * HW;
    const float* ib = inter + (size_t)b * CDIM * HW;
    float* swv = sh[wave];

    // B fragments from inter (K = input channel)
    v16bf bin[2];
#pragma unroll
    for (int kb = 0; kb < 2; ++kb) {
        const float* bp = ib + (size_t)(kb * 32 + csub) * HW + pix;
        float tmp[16];
#pragma unroll
        for (int j = 0; j < 16; ++j)
            tmp[j] = bp[(size_t)j * HW];
#pragma unroll
        for (int j = 0; j < 16; ++j)
            bin[kb][j] = (__bf16)tmp[j];
    }

    // gamma branch: h = lrelu(g1 @ inter); gamma = g2 @ h
    v8f hacc[4] = {};
    gemm64(pg1, bin, lane, hacc);
    store_h_lrelu(swv, hacc, lane);
    asm volatile("s_wait_dscnt 0" ::: "memory");              // wave-local LDS fence
    v16bf bh[2];
    load_hfrag(swv, bh, lane);
    v8f gacc[4] = {};
    gemm64(pg2, bh, lane, gacc);
    asm volatile("s_wait_dscnt 0" ::: "memory");

    // beta branch (reuses bin and the LDS region)
    v8f h2[4] = {};
    gemm64(pb1, bin, lane, h2);
    store_h_lrelu(swv, h2, lane);
    asm volatile("s_wait_dscnt 0" ::: "memory");
    load_hfrag(swv, bh, lane);
    v8f bacc[4] = {};
    gemm64(pb2, bh, lane, bacc);

    // final: out = x + dcn + x*gamma + beta   (dcn already in out)
    float* outb = out + (size_t)b * CDIM * HW;
#pragma unroll
    for (int mt = 0; mt < 4; ++mt)
#pragma unroll
        for (int r = 0; r < 8; ++r) {
            const int o = mt * 16 + (hi ? 8 + r : r);
            const size_t idx = (size_t)o * HW + pix;
            const float xv = xb[idx];
            const float d  = outb[idx];
            outb[idx] = xv + d + xv * gacc[mt][r] + bacc[mt][r];
        }
}

// ---------------------------------------------------------------------------
extern "C" void kernel_launch(void* const* d_in, const int* in_sizes, int n_in,
                              void* d_out, int out_size, void* d_ws, size_t ws_size,
                              hipStream_t stream) {
    const float* x     = (const float*)d_in[0];
    const float* inter = (const float*)d_in[1];
    const float* w_dcn = (const float*)d_in[2];
    const float* w_off = (const float*)d_in[3];
    const float* b_off = (const float*)d_in[4];
    const float* g1    = (const float*)d_in[5];
    const float* g2    = (const float*)d_in[6];
    const float* b1    = (const float*)d_in[7];
    const float* b2    = (const float*)d_in[8];
    float* out = (float*)d_out;

    char* ws = (char*)d_ws;
    float*  offbuf = (float*)ws;                              // 4*27*16384*4 = 7,077,888 B
    __bf16* pw_off = (__bf16*)(ws + 7077888);                 // 73,728 B
    __bf16* pw_dcn = (__bf16*)(ws + 7151616);                 // 73,728 B
    __bf16* pg1    = (__bf16*)(ws + 7225344);                 // 8,192 B each
    __bf16* pg2    = (__bf16*)(ws + 7233536);
    __bf16* pb1    = (__bf16*)(ws + 7241728);
    __bf16* pb2    = (__bf16*)(ws + 7249920);

    k_pack<<<64, 256, 0, stream>>>(w_off, w_dcn, g1, g2, b1, b2,
                                   pw_off, pw_dcn, pg1, pg2, pb1, pb2);

    dim3 grid(HW / 128, 4);                                   // 128 pixels per 8-wave block
    k_off_conv<<<grid, 256, 0, stream>>>(x, inter, pw_off, b_off, offbuf);
    k_dcn<<<grid, 256, 0, stream>>>(x, offbuf, pw_dcn, out);
    k_sft_final<<<grid, 256, 0, stream>>>(x, inter, pg1, pg2, pb1, pb2, out);
}
